// HybridGNN_78245714198919
// MI455X (gfx1250) — matrix operand order repaired and checked by reading, not compiled, not run on hardware
//
#include <hip/hip_runtime.h>
#include <hip/hip_bf16.h>
#include <cstdint>
#include <cstddef>

#define NUM_USERS 50000
#define NUM_ITEMS 20000
#define N_NODES   70000
#define BATCH_N   8192
#define NEDGES    500000
#define ETOT      (NEDGES + N_NODES)   /* edges + self loops */
#define NEG_SLOPE 0.2f

typedef __attribute__((ext_vector_type(16))) __bf16 v16bf;
typedef __attribute__((ext_vector_type(2)))  __bf16 v2bf;
typedef __attribute__((ext_vector_type(8)))  float  v8f;

union BF16x16 { v16bf v; unsigned short u[16]; uint4 q[2]; };

__device__ __forceinline__ unsigned short f32_bf16(float f) {
  unsigned int u = __float_as_uint(f);
  u += 0x7fffu + ((u >> 16) & 1u);           // round-to-nearest-even
  return (unsigned short)(u >> 16);
}

__device__ __forceinline__ unsigned int pack_bf16x2(float a, float b) {
#if __has_builtin(__builtin_amdgcn_cvt_pk_bf16_f32)
  v2bf p = __builtin_amdgcn_cvt_pk_bf16_f32(a, b);
  return __builtin_bit_cast(unsigned int, p);
#else
  return (unsigned int)f32_bf16(a) | ((unsigned int)f32_bf16(b) << 16);
#endif
}

// monotonic float<->uint mapping for atomicMax on floats
__device__ __forceinline__ unsigned int fflip(float f) {
  unsigned int u = __float_as_uint(f);
  return (u & 0x80000000u) ? ~u : (u | 0x80000000u);
}
__device__ __forceinline__ float funflip(unsigned int u) {
  u = (u & 0x80000000u) ? (u & 0x7fffffffu) : ~u;
  return __uint_as_float(u);
}

// ---------------------------------------------------------------------------
// C[M,N] = act( A[M,K] * W[N,K]^T + bias ).  A, W pre-converted to bf16.
// One wave computes a 16x64 strip. Per K-step, the A fragment and all four
// B fragments are fetched first (one clause of b128 loads, single wait),
// then 4 WMMAs issue back-to-back, overlapping the next iteration's loads.
// ---------------------------------------------------------------------------
__global__ void __launch_bounds__(128)
wmma_gemm(const unsigned short* __restrict__ A, const unsigned short* __restrict__ W,
          const float* __restrict__ bias, float* __restrict__ C,
          int M, int N, int K, int relu)
{
  const int lane  = threadIdx.x & 31;
  const int wave  = threadIdx.x >> 5;
  const int mTile = blockIdx.y * 4 + wave;
  if (mTile * 16 >= M) return;
  const int m0   = mTile * 16;
  const int n0   = blockIdx.x * 64;
  const int row  = lane & 15;
  const int half = lane >> 4;

  v8f acc[4] = {};
  const unsigned short* Arow = A + (size_t)(m0 + row) * K;

#pragma unroll 2
  for (int k0 = 0; k0 < K; k0 += 32) {
    __builtin_prefetch(Arow + k0 + 256, 0, 1);   // global_prefetch_b8
    // A fragment (16x32 bf16): elems 0-7 -> K=k0+half*8.., elems 8-15 -> +16
    BF16x16 af, bfr[4];
    af.q[0] = *(const uint4*)(Arow + k0 + half * 8);
    af.q[1] = *(const uint4*)(Arow + k0 + 16 + half * 8);
#pragma unroll
    for (int t = 0; t < 4; ++t) {
      // B fragment (32x16 bf16): lane = column, elems 0-7 -> K=k0+half*8.., elems 8-15 -> +16
      const unsigned short* Wrow =
          W + (size_t)(n0 + t * 16 + row) * K + k0 + half * 8;
      bfr[t].q[0] = *(const uint4*)(Wrow);
      bfr[t].q[1] = *(const uint4*)(Wrow + 16);
    }
#pragma unroll
    for (int t = 0; t < 4; ++t)
      acc[t] = __builtin_amdgcn_wmma_f32_16x16x32_bf16(
          false, af.v, false, bfr[t].v, (short)0, acc[t], false, false);
  }

#pragma unroll
  for (int t = 0; t < 4; ++t) {
    int n = n0 + t * 16 + row;
    float bv = bias ? bias[n] : 0.0f;
#pragma unroll
    for (int j = 0; j < 8; ++j) {          // D layout: m = m0 + half*8 + j
      float v = acc[t][j] + bv;
      if (relu) v = fmaxf(v, 0.0f);
      C[(size_t)(m0 + half * 8 + j) * N + n] = v;
    }
  }
}

// ---------------------------------------------------------------------------
__global__ void zero_u32(unsigned int* __restrict__ p, size_t n) {
  size_t i = (size_t)blockIdx.x * blockDim.x + threadIdx.x;
  size_t stride = (size_t)gridDim.x * blockDim.x;
  for (; i < n; i += stride) p[i] = 0u;
}

// packed f32 -> bf16 conversion (n must be even)
__global__ void cvt_f32_bf16(const float* __restrict__ src,
                             unsigned short* __restrict__ dst, size_t n)
{
  size_t i = ((size_t)blockIdx.x * blockDim.x + threadIdx.x) * 2;
  size_t stride = (size_t)gridDim.x * blockDim.x * 2;
  for (; i < n; i += stride)
    *(unsigned int*)(dst + i) = pack_bf16x2(src[i], src[i + 1]);
}

// scatter batch rows into sparse bf16 node matrix x[N_NODES,256]
__global__ void scatter_nodes(const int* __restrict__ user_ids,
                              const int* __restrict__ item_ids,
                              const float* __restrict__ user_emb,
                              const float* __restrict__ item_emb,
                              const float* __restrict__ cemb,
                              unsigned short* __restrict__ xb)
{
  int b = blockIdx.x;            // batch element
  int t = threadIdx.x * 2;       // 0..126 step 2 (64 threads)
  int u  = user_ids[b];
  int it = item_ids[b];
  *(unsigned int*)(xb + (size_t)u * 256 + t) =
      pack_bf16x2(user_emb[(size_t)u * 128 + t], user_emb[(size_t)u * 128 + t + 1]);
  *(unsigned int*)(xb + (size_t)(NUM_USERS + it) * 256 + t) =
      pack_bf16x2(item_emb[(size_t)it * 128 + t], item_emb[(size_t)it * 128 + t + 1]);
  *(unsigned int*)(xb + (size_t)(NUM_USERS + it) * 256 + 128 + t) =
      pack_bf16x2(cemb[(size_t)b * 128 + t], cemb[(size_t)b * 128 + t + 1]);
}

// per-(node,head) attention logits: wave reduction over 128 channels
__global__ void node_att(const float* __restrict__ h,
                         const float* __restrict__ att_src,
                         const float* __restrict__ att_dst,
                         float* __restrict__ a_src, float* __restrict__ a_dst,
                         int Nn, int H)
{
  int lane = threadIdx.x & 31;
  int wid  = blockIdx.x * (blockDim.x >> 5) + (threadIdx.x >> 5);
  if (wid >= Nn * H) return;
  int n = wid / H, hh = wid % H;
  const float* hp = h + (size_t)n * (H * 128) + hh * 128;
  const float* as = att_src + hh * 128;
  const float* ad = att_dst + hh * 128;
  float s = 0.f, d = 0.f;
  for (int c = lane; c < 128; c += 32) { float v = hp[c]; s += v * as[c]; d += v * ad[c]; }
  for (int o = 16; o; o >>= 1) { s += __shfl_xor(s, o); d += __shfl_xor(d, o); }
  if (lane == 0) { a_src[(size_t)n * H + hh] = s; a_dst[(size_t)n * H + hh] = d; }
}

// pass 1: per-edge leaky_relu logit + segment max via flipped-uint atomicMax
__global__ void edge_alpha_max(const int* __restrict__ eidx,
                               const float* __restrict__ a_src,
                               const float* __restrict__ a_dst,
                               float* __restrict__ alpha,
                               unsigned int* __restrict__ amax,
                               int E, int Etot, int H)
{
  int idx = blockIdx.x * blockDim.x + threadIdx.x;
  if (idx >= Etot * H) return;
  int e = idx / H, hh = idx % H;
  int s = (e < E) ? eidx[e]     : (e - E);     // self loop past E
  int d = (e < E) ? eidx[E + e] : (e - E);
  float a = a_src[(size_t)s * H + hh] + a_dst[(size_t)d * H + hh];
  a = (a > 0.f) ? a : a * NEG_SLOPE;
  alpha[idx] = a;
  atomicMax(&amax[(size_t)d * H + hh], fflip(a));
}

// pass 2 (fused): denom += ex ; agg[dst] += ex * h[src]  (normalize later)
__global__ void edge_accum(const int* __restrict__ eidx,
                           const float* __restrict__ alpha,
                           const unsigned int* __restrict__ amax,
                           const float* __restrict__ h,
                           float* __restrict__ denom,
                           float* __restrict__ agg,
                           int E, int Etot, int H)
{
  int lane = threadIdx.x & 31;
  int e = blockIdx.x * (blockDim.x >> 5) + (threadIdx.x >> 5);
  if (e >= Etot) return;
  int s = (e < E) ? eidx[e]     : (e - E);
  int d = (e < E) ? eidx[E + e] : (e - E);
  for (int hh = 0; hh < H; ++hh) {
    float ex = __expf(alpha[(size_t)e * H + hh] - funflip(amax[(size_t)d * H + hh]));
    if (lane == 0) atomicAdd(&denom[(size_t)d * H + hh], ex);
    const float4* hs = (const float4*)(h + (size_t)s * (H * 128) + hh * 128);
    float*        ag = agg + (size_t)d * (H * 128) + hh * 128;
    float4 hv = hs[lane];                     // coalesced b128 load
    atomicAdd(&ag[lane * 4 + 0], ex * hv.x);
    atomicAdd(&ag[lane * 4 + 1], ex * hv.y);
    atomicAdd(&ag[lane * 4 + 2], ex * hv.z);
    atomicAdd(&ag[lane * 4 + 3], ex * hv.w);
  }
}

// normalize + bias (+ optional ELU); writes f32 and/or bf16
__global__ void finalize(const float* __restrict__ agg, const float* __restrict__ denom,
                         const float* __restrict__ bias,
                         float* __restrict__ outf, unsigned short* __restrict__ outb,
                         int Nn, int H, int elu)
{
  int idx = (blockIdx.x * blockDim.x + threadIdx.x) * 2;   // pair of channels
  if (idx >= Nn * H * 128) return;
  int n  = idx / (H * 128);
  int f  = idx % (H * 128);
  int hh = f >> 7;
  float dinv = 1.0f / denom[(size_t)n * H + hh];
  float v0 = agg[idx]     * dinv + bias[f];
  float v1 = agg[idx + 1] * dinv + bias[f + 1];
  if (elu) {
    v0 = (v0 > 0.f) ? v0 : (__expf(v0) - 1.0f);
    v1 = (v1 > 0.f) ? v1 : (__expf(v1) - 1.0f);
  }
  if (outb) *(unsigned int*)(outb + idx) = pack_bf16x2(v0, v1);
  if (outf) { outf[idx] = v0; outf[idx + 1] = v1; }
}

// combined = [u_emb | x_out[item node] | x_out[user node]]  (B x 384, bf16)
__global__ void gather_combined(const int* __restrict__ user_ids,
                                const int* __restrict__ item_ids,
                                const float* __restrict__ user_emb,
                                const float* __restrict__ xo,
                                unsigned short* __restrict__ comb)
{
  int b = blockIdx.x, t = threadIdx.x * 2;   // 64 threads, pairs
  int u = user_ids[b], it = item_ids[b];
  *(unsigned int*)(comb + (size_t)b * 384 + t) =
      pack_bf16x2(user_emb[(size_t)u * 128 + t], user_emb[(size_t)u * 128 + t + 1]);
  *(unsigned int*)(comb + (size_t)b * 384 + 128 + t) =
      pack_bf16x2(xo[(size_t)(NUM_USERS + it) * 128 + t],
                  xo[(size_t)(NUM_USERS + it) * 128 + t + 1]);
  *(unsigned int*)(comb + (size_t)b * 384 + 256 + t) =
      pack_bf16x2(xo[(size_t)u * 128 + t], xo[(size_t)u * 128 + t + 1]);
}

// final dot with pred_w2: one wave per batch element
__global__ void pred_out(const float* __restrict__ hp,
                         const float* __restrict__ w2, const float* __restrict__ b2,
                         float* __restrict__ out)
{
  int lane = threadIdx.x & 31;
  int b = blockIdx.x * (blockDim.x >> 5) + (threadIdx.x >> 5);
  if (b >= BATCH_N) return;
  const float* hb = hp + (size_t)b * 128;
  float s = 0.f;
  for (int c = lane; c < 128; c += 32) s += hb[c] * w2[c];
  for (int o = 16; o; o >>= 1) s += __shfl_xor(s, o);
  if (lane == 0) out[b] = s + b2[0];
}

// ---------------------------------------------------------------------------
extern "C" void kernel_launch(void* const* d_in, const int* in_sizes, int n_in,
                              void* d_out, int out_size, void* d_ws, size_t ws_size,
                              hipStream_t stream)
{
  const int*   user_ids = (const int*)d_in[0];
  const int*   item_ids = (const int*)d_in[1];
  const float* content  = (const float*)d_in[2];
  const int*   eidx     = (const int*)d_in[3];
  const float* user_emb = (const float*)d_in[4];
  const float* item_emb = (const float*)d_in[5];
  const float* cp_w     = (const float*)d_in[6];
  const float* cp_b     = (const float*)d_in[7];
  const float* c1_w     = (const float*)d_in[8];
  const float* c1_as    = (const float*)d_in[9];
  const float* c1_ad    = (const float*)d_in[10];
  const float* c1_b     = (const float*)d_in[11];
  const float* c2_w     = (const float*)d_in[12];
  const float* c2_as    = (const float*)d_in[13];
  const float* c2_ad    = (const float*)d_in[14];
  const float* c2_b     = (const float*)d_in[15];
  const float* p_w1     = (const float*)d_in[16];
  const float* p_b1     = (const float*)d_in[17];
  const float* p_w2     = (const float*)d_in[18];
  const float* p_b2     = (const float*)d_in[19];
  float* out = (float*)d_out;
  (void)in_sizes; (void)n_in; (void)out_size; (void)ws_size;

  char* base = (char*)d_ws;
  size_t off = 0;
  auto alloc = [&](size_t bytes) -> char* {
    char* p = base + off;
    off = (off + bytes + 255) & ~(size_t)255;
    return p;
  };
  unsigned short* xb   = (unsigned short*)alloc((size_t)N_NODES * 256 * 2); // x1 -> elu(x1)
  float*          h    = (float*)alloc((size_t)N_NODES * 256 * 4);          // h1 -> h2
  float*          agg  = (float*)alloc((size_t)N_NODES * 256 * 4);
  float*          xo   = (float*)alloc((size_t)N_NODES * 128 * 4);          // conv2 out
  unsigned short* cb   = (unsigned short*)alloc((size_t)BATCH_N * 512 * 2);
  float*          cemb = (float*)alloc((size_t)BATCH_N * 128 * 4);
  unsigned short* cpwb = (unsigned short*)alloc((size_t)128 * 512 * 2);
  unsigned short* c1wb = (unsigned short*)alloc((size_t)256 * 256 * 2);
  unsigned short* c2wb = (unsigned short*)alloc((size_t)128 * 256 * 2);
  unsigned short* pw1b = (unsigned short*)alloc((size_t)128 * 384 * 2);
  float*          asrc = (float*)alloc((size_t)N_NODES * 2 * 4);
  float*          adst = (float*)alloc((size_t)N_NODES * 2 * 4);
  float*          alph = (float*)alloc((size_t)ETOT * 2 * 4);
  unsigned int*   amax = (unsigned int*)alloc((size_t)N_NODES * 2 * 4);
  float*          denom= (float*)alloc((size_t)N_NODES * 2 * 4);
  unsigned short* combb= (unsigned short*)alloc((size_t)BATCH_N * 384 * 2);
  float*          ph   = (float*)alloc((size_t)BATCH_N * 128 * 4);

  auto gemm = [&](const unsigned short* A, const unsigned short* W,
                  const float* bias, float* C, int M, int N, int K, int relu) {
    dim3 grid(N / 64, ((M + 15) / 16 + 3) / 4);
    wmma_gemm<<<grid, 128, 0, stream>>>(A, W, bias, C, M, N, K, relu);
  };
  auto cvt = [&](const float* s, unsigned short* d, size_t n) {
    cvt_f32_bf16<<<1024, 256, 0, stream>>>(s, d, n);
  };

  auto conv_att = [&](const float* hbuf, const float* a_s, const float* a_d,
                      const float* bias, int H, int elu,
                      float* outf, unsigned short* outb) {
    zero_u32<<<2048, 256, 0, stream>>>((unsigned int*)agg, (size_t)N_NODES * H * 128);
    zero_u32<<<128, 256, 0, stream>>>((unsigned int*)denom, (size_t)N_NODES * H);
    zero_u32<<<128, 256, 0, stream>>>(amax, (size_t)N_NODES * H);
    int NH = N_NODES * H;
    node_att<<<(NH + 7) / 8, 256, 0, stream>>>(hbuf, a_s, a_d, asrc, adst, N_NODES, H);
    int EH = ETOT * H;
    edge_alpha_max<<<(EH + 255) / 256, 256, 0, stream>>>(eidx, asrc, adst, alph, amax,
                                                         NEDGES, ETOT, H);
    edge_accum<<<(ETOT + 7) / 8, 256, 0, stream>>>(eidx, alph, amax, hbuf, denom, agg,
                                                   NEDGES, ETOT, H);
    int pairs = N_NODES * H * 64;
    finalize<<<(pairs + 255) / 256, 256, 0, stream>>>(agg, denom, bias, outf, outb,
                                                      N_NODES, H, elu);
  };

  // 0) one-shot bf16 conversions (weights + content)
  cvt(content, cb, (size_t)BATCH_N * 512);
  cvt(cp_w, cpwb, (size_t)128 * 512);
  cvt(c1_w, c1wb, (size_t)256 * 256);
  cvt(c2_w, c2wb, (size_t)128 * 256);
  cvt(p_w1, pw1b, (size_t)128 * 384);
  // 1) zero bf16 node matrix, project content, scatter batch rows (bf16)
  zero_u32<<<2048, 256, 0, stream>>>((unsigned int*)xb, (size_t)N_NODES * 128);
  gemm(cb, cpwb, cp_b, cemb, BATCH_N, 128, 512, 0);
  scatter_nodes<<<BATCH_N, 64, 0, stream>>>(user_ids, item_ids, user_emb, item_emb,
                                            cemb, xb);
  // 2) GAT conv1 (heads=2) + ELU -> bf16 directly
  gemm(xb, c1wb, nullptr, h, N_NODES, 256, 256, 0);
  conv_att(h, c1_as, c1_ad, c1_b, 2, /*elu=*/1, nullptr, xb);
  // 3) GAT conv2 (heads=1) -> f32
  gemm(xb, c2wb, nullptr, h, N_NODES, 128, 256, 0);
  conv_att(h, c2_as, c2_ad, c2_b, 1, /*elu=*/0, xo, nullptr);
  // 4) prediction head
  gather_combined<<<BATCH_N, 64, 0, stream>>>(user_ids, item_ids, user_emb, xo, combb);
  gemm(combb, pw1b, p_b1, ph, BATCH_N, 128, 384, /*relu=*/1);
  pred_out<<<(BATCH_N + 7) / 8, 256, 0, stream>>>(ph, p_w2, p_b2, out);
}
